// MemModule_84739704750653
// MI455X (gfx1250) — matrix operand order repaired and checked by reading, not compiled, not run on hardware
//
#include <hip/hip_runtime.h>
#include <hip/hip_bf16.h>

// ---------------------------------------------------------------------------
// MemAE memory module, fused for MI455X (gfx1250, wave32, WMMA bf16).
//   att = softmax(q @ M^T); att = shrink(att); att /= L1; y = att @ M
// One workgroup = 16 consecutive pixels (same batch image, contiguous hw),
// which makes both att_map ([B,mem,H,W]) and y ([B,C,H,W]) writes coalesced.
// memMatrix is pre-packed into WMMA B-fragment order (bf16, padded to 2048)
// so every B operand is a contiguous 32B/lane global load out of L2.
// ---------------------------------------------------------------------------

typedef __attribute__((ext_vector_type(16))) __bf16 v16bf;
typedef __attribute__((ext_vector_type(8)))  __bf16 v8bf;
typedef __attribute__((ext_vector_type(8)))  float  v8f;
typedef __attribute__((ext_vector_type(4)))  float  v4f;

#define MEM_DIM   2000
#define MEM_PAD   2048
#define FEA       256
#define HW        1024          // 32*32
#define LAMBDA    0.0025f
#define EPS_SHR   1e-12f
#define EPS_NRM   1e-12f

#define QS 264                  // lds q row stride (bf16)
#define SS 2052                 // lds scores row stride (f32)
#define OS 260                  // lds out row stride (f32)

__device__ __forceinline__ __bf16 f2bf(float f) {
    unsigned u = __builtin_bit_cast(unsigned, f);
    u += 0x7FFFu + ((u >> 16) & 1u);          // round-to-nearest-even
    unsigned short h = (unsigned short)(u >> 16);
    return __builtin_bit_cast(__bf16, h);
}

// --- pre-pass: pack memMatrix into WMMA B-fragment layouts (bf16, padded) ---
// mb1: GEMM1 B = M^T tiles: frag = nTile(mem/16)*8 + kTile(fea/32); 512 elem/frag
// mb2: GEMM2 B = M  tiles:  frag = kTile(mem/32)*16 + nTile(fea/16)
// B-frag lane layout (16-bit, 32x16): n = lane&15, K = (e>>3)*16 + (lane>>4)*8 + (e&7)
__global__ void pack_mem_kernel(const float* __restrict__ M,
                                __bf16* __restrict__ mb1,
                                __bf16* __restrict__ mb2) {
    const unsigned gid = blockIdx.x * 256u + threadIdx.x;   // < 524288
    const int e    = gid & 15;
    const int lane = (gid >> 4) & 31;
    const unsigned frag = gid >> 9;
    {   // GEMM1: B[k=fea][n=mem] = M[mem][fea]
        const int kTile = frag & 7, nTile = frag >> 3;
        const int j = nTile * 16 + (lane & 15);               // mem row
        const int k = kTile * 32 + (e >> 3) * 16 + (lane >> 4) * 8 + (e & 7);      // fea
        mb1[gid] = f2bf(j < MEM_DIM ? M[j * FEA + k] : 0.f);
    }
    {   // GEMM2: B[k=mem][n=fea] = M[mem][fea]
        const int nTile = frag & 15, kTile = frag >> 4;
        const int j = kTile * 32 + (e >> 3) * 16 + (lane >> 4) * 8 + (e & 7);      // mem row
        const int c = nTile * 16 + (lane & 15);               // fea
        mb2[gid] = f2bf(j < MEM_DIM ? M[j * FEA + c] : 0.f);
    }
}

// --------------------------- fused main kernel ------------------------------
__global__ __launch_bounds__(256)
void memae_fused_kernel(const float*  __restrict__ x,
                        const __bf16* __restrict__ mb1,
                        const __bf16* __restrict__ mb2,
                        float* __restrict__ y,
                        float* __restrict__ att) {
    __shared__ __align__(16) unsigned char smem[140864];
    __bf16* q      = (__bf16*)smem;                 // [16][QS]  bf16   8448 B
    float*  sc     = (float*)(smem + 8448);         // [16][SS]  f32  131328 B
    float*  red    = (float*)(smem + 139776);       // [16][16]  f32
    float*  rowInv = (float*)(smem + 140800);       // [16]      f32
    float*  outT   = (float*)(smem + 8448);         // [16][OS]  aliases sc

    const int tid  = threadIdx.x;
    const int wave = tid >> 5;          // 0..7
    const int lane = tid & 31;
    const int s    = blockIdx.x;        // 0..2047
    const int b    = s >> 6;            // image index
    const int hw0  = (s & 63) * 16;     // first pixel (contiguous 16)
    const size_t xb = (size_t)b * FEA * HW;

    // ---- stage q tile: 16 pixels x 256 features, fp32 -> bf16 ----
    for (int idx = tid; idx < 16 * FEA; idx += 256) {
        const int c = idx >> 4, p = idx & 15;
        q[p * QS + c] = f2bf(x[xb + (size_t)c * HW + hw0 + p]);
    }
    __syncthreads();

    union AFrag { v16bf v; v8bf h[2]; __bf16 e[16]; };
    const int row = lane & 15;          // A row / C col index helpers
    const int k8  = (lane >> 4) * 8;    // A K sub-offset (hi half-wave)
    const int rhi = (lane >> 4) * 8;    // C row offset (hi half-wave)

    // ---- GEMM1: scores[16 x 2048] = q @ M^T ----
    {
        AFrag a1[8];                    // hoisted A fragments (K = 256)
        const __bf16* qr = q + row * QS;
#pragma unroll
        for (int kt = 0; kt < 8; ++kt) {        // A 16-bit layout: two 16B runs
            a1[kt].h[0] = *(const v8bf*)(qr + kt * 32 + k8);
            a1[kt].h[1] = *(const v8bf*)(qr + kt * 32 + 16 + k8);
        }
        for (int t = 0; t < 16; ++t) {          // 16 mem-tiles per wave
            const int nTile = wave * 16 + t;
            const __bf16* bp = mb1 + (size_t)nTile * 8 * 512 + lane * 16;
            v8f acc = {};
#pragma unroll
            for (int kt = 0; kt < 8; ++kt) {
                v16bf bf = *(const v16bf*)(bp + kt * 512);
                acc = __builtin_amdgcn_wmma_f32_16x16x32_bf16(
                        false, a1[kt].v, false, bf, (short)0, acc, false, false);
            }
#pragma unroll
            for (int v = 0; v < 8; ++v)
                sc[(v + rhi) * SS + nTile * 16 + row] = acc[v];
        }
    }
    __syncthreads();

    // ---- softmax + hard-shrink (16 threads per pixel row) ----
    {
        const int r = tid >> 4, sub = tid & 15;
        float* srow = sc + r * SS;
        float m = -3.0e38f;
        for (int j = sub; j < MEM_DIM; j += 16) m = fmaxf(m, srow[j]);
        red[r * 16 + sub] = m;
        __syncthreads();
#pragma unroll
        for (int i = 0; i < 16; ++i) m = fmaxf(m, red[r * 16 + i]);
        float se = 0.f;
        for (int j = sub; j < MEM_DIM; j += 16) {
            const float e = __expf(srow[j] - m);
            srow[j] = e;
            se += e;
        }
        __syncthreads();
        red[r * 16 + sub] = se;
        __syncthreads();
        se = 0.f;
#pragma unroll
        for (int i = 0; i < 16; ++i) se += red[r * 16 + i];
        const float invSe = 1.f / se;
        float sh = 0.f;
        for (int j = sub; j < MEM_DIM; j += 16) {   // p -> shrink(p) = h
            const float p = srow[j] * invSe;
            const float d = p - LAMBDA;
            const float h = fmaxf(d, 0.f) * p / (fabsf(d) + EPS_SHR);
            srow[j] = h;
            sh += h;
        }
        __syncthreads();
        red[r * 16 + sub] = sh;
        __syncthreads();
        sh = 0.f;
#pragma unroll
        for (int i = 0; i < 16; ++i) sh += red[r * 16 + i];
        if (sub == 0) rowInv[r] = 1.f / fmaxf(sh, EPS_NRM);
        __syncthreads();
    }

    // ---- att_map write: j-major so 16 lanes store 64B contiguous ----
    {
        const int g = tid >> 4, i = tid & 15;
        const float* si  = sc + i * SS;
        const float invD = rowInv[i];
        const size_t ab  = (size_t)b * MEM_DIM * HW + hw0 + i;
        for (int j = g; j < MEM_DIM; j += 16)
            att[ab + (size_t)j * HW] = si[j] * invD;
    }

    // ---- GEMM2: out[16 x 256] = h @ M, L1-renorm folded into epilogue ----
    v8f acc0 = {}, acc1 = {};
    {
        const float* srow2 = sc + row * SS;
        const __bf16* bb0 = mb2 + (size_t)(2 * wave)     * 512 + lane * 16;
        const __bf16* bb1 = mb2 + (size_t)(2 * wave + 1) * 512 + lane * 16;
        for (int kt = 0; kt < 64; ++kt) {
            const float* p0 = srow2 + kt * 32 + k8;
            v4f f0 = *(const v4f*)(p0);
            v4f f1 = *(const v4f*)(p0 + 4);
            v4f f2 = *(const v4f*)(p0 + 16);
            v4f f3 = *(const v4f*)(p0 + 20);
            AFrag a;
#pragma unroll
            for (int e = 0; e < 4; ++e) {
                a.e[e]      = f2bf(f0[e]);
                a.e[4 + e]  = f2bf(f1[e]);
                a.e[8 + e]  = f2bf(f2[e]);
                a.e[12 + e] = f2bf(f3[e]);
            }
            v16bf b0 = *(const v16bf*)(bb0 + (size_t)kt * 8192);
            v16bf b1 = *(const v16bf*)(bb1 + (size_t)kt * 8192);
            acc0 = __builtin_amdgcn_wmma_f32_16x16x32_bf16(
                     false, a.v, false, b0, (short)0, acc0, false, false);
            acc1 = __builtin_amdgcn_wmma_f32_16x16x32_bf16(
                     false, a.v, false, b1, (short)0, acc1, false, false);
        }
    }
    float inv[8];
#pragma unroll
    for (int v = 0; v < 8; ++v) inv[v] = rowInv[v + rhi];
    __syncthreads();                      // scores dead; outT may alias them
#pragma unroll
    for (int v = 0; v < 8; ++v) {
        outT[(v + rhi) * OS + wave * 32 + row]      = acc0[v] * inv[v];
        outT[(v + rhi) * OS + wave * 32 + 16 + row] = acc1[v] * inv[v];
    }
    __syncthreads();
    // ---- y write: coalesced 16-float runs per channel ----
    {
        const int g = tid >> 4, i = tid & 15;
        for (int c = g; c < FEA; c += 16)
            y[xb + (size_t)c * HW + hw0 + i] = outT[i * OS + c];
    }
}

extern "C" void kernel_launch(void* const* d_in, const int* in_sizes, int n_in,
                              void* d_out, int out_size, void* d_ws, size_t ws_size,
                              hipStream_t stream) {
    const float* x = (const float*)d_in[0];          // [32,256,32,32]
    const float* M = (const float*)d_in[1];          // [2000,256]
    float* y   = (float*)d_out;                      // [32,256,32,32]
    float* att = (float*)d_out + (size_t)32 * FEA * HW; // [32,2000,32,32]

    __bf16* mb1 = (__bf16*)d_ws;                     // 524288 bf16 = 1 MB
    __bf16* mb2 = mb1 + 524288;                      // 524288 bf16 = 1 MB

    pack_mem_kernel<<<2048, 256, 0, stream>>>(M, mb1, mb2);
    memae_fused_kernel<<<2048, 256, 0, stream>>>(x, mb1, mb2, y, att);
}
